// TopologicalHotspotGNN_2920577761627
// MI455X (gfx1250) — matrix operand order repaired and checked by reading, not compiled, not run on hardware
//
#include <hip/hip_runtime.h>
#include <hip/hip_bf16.h>

typedef __attribute__((ext_vector_type(16))) _Float16 v16h;
typedef __attribute__((ext_vector_type(8)))  float    v8f;

// ---------------- ordered-float helpers for atomic max on floats ----------
__device__ __forceinline__ unsigned ordf(float f) {
    unsigned u = __float_as_uint(f);
    return (u & 0x80000000u) ? ~u : (u | 0x80000000u);
}
__device__ __forceinline__ float unordf(unsigned u) {
    unsigned b = (u & 0x80000000u) ? (u & 0x7fffffffu) : ~u;
    return __uint_as_float(b);
}

// ---------------- WMMA GEMM: out[N,64] = in[N,64] @ Wg[64,64] -------------
// 256 threads = 8 waves; each wave owns a 16-row stripe and all 4 col tiles.
__global__ __launch_bounds__(256)
void gemm64_wmma(const float* __restrict__ in, const float* __restrict__ Wg,
                 float* __restrict__ out, int nrows) {
    __shared__ __align__(32) _Float16 WT[64 * 64];   // [col n][k], transposed
    int t = threadIdx.x;
#pragma unroll
    for (int i = 0; i < 16; ++i) {
        int idx = t + i * 256;
        int n = idx >> 6, k = idx & 63;
        WT[idx] = (_Float16)Wg[k * 64 + n];
    }
    __syncthreads();

    int wave = t >> 5, lane = t & 31;
    int rt = blockIdx.x * 8 + wave;                  // 16-row tile index
    int nRowTiles = (nrows + 15) >> 4;
    if (rt >= nRowTiles) return;

    int r = lane & 15, g = lane >> 4;
    int row = rt * 16 + r;
    int rowc = row < nrows ? row : nrows - 1;        // clamp for tail loads
    const float* p = in + (long long)rowc * 64;

    // A fragments for k-step 0 (K 0..31) and 1 (K 32..63).
    // lane-group g: halves 0..15 -> K = ks*32 + g*16 + i
    v16h a[2];
#pragma unroll
    for (int ks = 0; ks < 2; ++ks) {
        int base = ks * 32 + g * 16;
#pragma unroll
        for (int i = 0; i < 8; ++i) {
            a[ks][i]     = (_Float16)p[base + i];
            a[ks][8 + i] = (_Float16)p[base + 8 + i];
        }
    }

#pragma unroll
    for (int ct = 0; ct < 4; ++ct) {
        v8f c = {};
#pragma unroll
        for (int ks = 0; ks < 2; ++ks) {
            // B fragment: lane col = r, group g covers K = ks*32 + g*16 + [0,16)
            const _Float16* bp = &WT[(ct * 16 + r) * 64 + ks * 32 + g * 16];
            v16h b = *(const v16h*)bp;               // 32B aligned LDS read
            c = __builtin_amdgcn_wmma_f32_16x16x32_f16(
                    false, a[ks], false, b, (short)0, c, false, false);
        }
        if (row < nrows) {
#pragma unroll
            for (int v = 0; v < 8; ++v)
                out[(long long)(rt * 16 + g * 8 + v) * 64 + ct * 16 + r] = c[v];
        } else {
            // rows of this tile may partially exist; store the valid ones
#pragma unroll
            for (int v = 0; v < 8; ++v) {
                int rr = rt * 16 + g * 8 + v;
                if (rr < nrows) out[(long long)rr * 64 + ct * 16 + r] = c[v];
            }
        }
    }
}

// ---------------- per-node attention coefficients + init ------------------
__global__ void node_prep_k(const float* __restrict__ h,
                            const float* __restrict__ as_w,
                            const float* __restrict__ ad_w,
                            float* __restrict__ a_src, float* __restrict__ a_dst,
                            unsigned* __restrict__ m_ord, float* __restrict__ s_sum,
                            int n) {
    int i = blockIdx.x * blockDim.x + threadIdx.x;
    if (i >= n) return;
    float as[4] = {0.f, 0.f, 0.f, 0.f}, ad[4] = {0.f, 0.f, 0.f, 0.f};
    const float* p = h + (long long)i * 64;
#pragma unroll
    for (int c = 0; c < 64; ++c) {
        float v = p[c];
        as[c >> 4] = fmaf(v, as_w[c], as[c >> 4]);
        ad[c >> 4] = fmaf(v, ad_w[c], ad[c >> 4]);
    }
#pragma unroll
    for (int hd = 0; hd < 4; ++hd) {
        a_src[i * 4 + hd] = as[hd];
        a_dst[i * 4 + hd] = ad[hd];
        m_ord[i * 4 + hd] = 0u;       // ordered-float -inf
        s_sum[i * 4 + hd] = 0.f;
    }
}

// ---------------- edge helpers -------------------------------------------
__device__ __forceinline__ void edge_ends(const long long* __restrict__ ei,
                                          long long E, long long e,
                                          int& src, int& dst) {
    if (e < E) { src = (int)ei[e]; dst = (int)ei[E + e]; }
    else       { src = dst = (int)(e - E); }          // appended self loops
}

__device__ __forceinline__ float leaky02(float x) {
    return x > 0.f ? x : 0.2f * x;
}

__global__ void edge_max_k(const long long* __restrict__ ei,
                           const float* __restrict__ a_src,
                           const float* __restrict__ a_dst,
                           unsigned* __restrict__ m_ord, long long E, int N) {
    long long t = (long long)blockIdx.x * blockDim.x + threadIdx.x;
    long long total = (E + N) * 4;
    if (t >= total) return;
    long long e = t >> 2; int hd = (int)(t & 3);
    int src, dst; edge_ends(ei, E, e, src, dst);
    float l = leaky02(a_src[src * 4 + hd] + a_dst[dst * 4 + hd]);
    atomicMax(&m_ord[dst * 4 + hd], ordf(l));
}

__global__ void edge_sum_k(const long long* __restrict__ ei,
                           const float* __restrict__ a_src,
                           const float* __restrict__ a_dst,
                           const unsigned* __restrict__ m_ord,
                           float* __restrict__ s_sum, long long E, int N) {
    long long t = (long long)blockIdx.x * blockDim.x + threadIdx.x;
    long long total = (E + N) * 4;
    if (t >= total) return;
    long long e = t >> 2; int hd = (int)(t & 3);
    int src, dst; edge_ends(ei, E, e, src, dst);
    float l = leaky02(a_src[src * 4 + hd] + a_dst[dst * 4 + hd]);
    atomicAdd(&s_sum[dst * 4 + hd], __expf(l - unordf(m_ord[dst * 4 + hd])));
}

__global__ void edge_agg_k(const long long* __restrict__ ei,
                           const float* __restrict__ a_src,
                           const float* __restrict__ a_dst,
                           const unsigned* __restrict__ m_ord,
                           const float* __restrict__ s_sum,
                           const float* __restrict__ h,
                           float* __restrict__ agg, long long E, int N) {
    long long t = (long long)blockIdx.x * blockDim.x + threadIdx.x;
    long long total = (E + N) * 64;
    if (t >= total) return;
    long long e = t >> 6; int c = (int)(t & 63); int hd = c >> 4;
    int src, dst; edge_ends(ei, E, e, src, dst);
    float l = leaky02(a_src[src * 4 + hd] + a_dst[dst * 4 + hd]);
    float alpha = __expf(l - unordf(m_ord[dst * 4 + hd])) /
                  (s_sum[dst * 4 + hd] + 1e-16f);
    atomicAdd(&agg[(long long)dst * 64 + c], alpha * h[(long long)src * 64 + c]);
}

// ---------------- misc elementwise ---------------------------------------
__global__ void zero_k(float* __restrict__ p, long long n) {
    long long t = (long long)blockIdx.x * blockDim.x + threadIdx.x;
    if (t < n) p[t] = 0.f;
}

__global__ void bias_elu_k(float* __restrict__ agg, const float* __restrict__ bias,
                           long long total) {
    long long t = (long long)blockIdx.x * blockDim.x + threadIdx.x;
    if (t >= total) return;
    float v = agg[t] + bias[t & 63];
    agg[t] = v > 0.f ? v : (__expf(v) - 1.f);
}

// ---------------- MLP head: 64 -> relu(32) -> sigmoid(1) ------------------
__global__ __launch_bounds__(256)
void mlp_head_k(const float* __restrict__ feat, const float* __restrict__ hW1,
                const float* __restrict__ hb1, const float* __restrict__ hW2,
                const float* __restrict__ hb2, float* __restrict__ out, int n) {
    __shared__ float W1s[64 * 32];
    __shared__ float b1s[32], W2s[32];
    int t = threadIdx.x;
    for (int i = t; i < 2048; i += 256) W1s[i] = hW1[i];
    if (t < 32) { b1s[t] = hb1[t]; W2s[t] = hW2[t]; }
    __syncthreads();
    int i = blockIdx.x * 256 + t;
    if (i >= n) return;
    float acc[32];
#pragma unroll
    for (int j = 0; j < 32; ++j) acc[j] = b1s[j];
    const float* f = feat + (long long)i * 64;
    for (int k = 0; k < 64; ++k) {
        float v = f[k];
#pragma unroll
        for (int j = 0; j < 32; ++j) acc[j] = fmaf(v, W1s[k * 32 + j], acc[j]);
    }
    float o = hb2[0];
#pragma unroll
    for (int j = 0; j < 32; ++j) o += fmaxf(acc[j], 0.f) * W2s[j];
    out[i] = 1.f / (1.f + __expf(-o));
}

// ---------------- host-side layer driver ----------------------------------
static void run_gat_layer(const float* in_feat, const float* W,
                          const float* att_s, const float* att_d, const float* bias,
                          const long long* ei, long long E, int N,
                          float* h_trans, float* agg,
                          float* a_src, float* a_dst, unsigned* m_ord, float* s_sum,
                          hipStream_t stream) {
    int nRowTiles = (N + 15) >> 4;
    int gemmBlocks = (nRowTiles + 7) / 8;
    gemm64_wmma<<<gemmBlocks, 256, 0, stream>>>(in_feat, W, h_trans, N);

    node_prep_k<<<(N + 255) / 256, 256, 0, stream>>>(
        h_trans, att_s, att_d, a_src, a_dst, m_ord, s_sum, N);

    long long nfeat = (long long)N * 64;
    zero_k<<<(unsigned)((nfeat + 255) / 256), 256, 0, stream>>>(agg, nfeat);

    long long t4 = (E + N) * 4;
    edge_max_k<<<(unsigned)((t4 + 255) / 256), 256, 0, stream>>>(
        ei, a_src, a_dst, m_ord, E, N);
    edge_sum_k<<<(unsigned)((t4 + 255) / 256), 256, 0, stream>>>(
        ei, a_src, a_dst, m_ord, s_sum, E, N);

    long long t64 = (E + N) * 64;
    edge_agg_k<<<(unsigned)((t64 + 255) / 256), 256, 0, stream>>>(
        ei, a_src, a_dst, m_ord, s_sum, h_trans, agg, E, N);

    bias_elu_k<<<(unsigned)((nfeat + 255) / 256), 256, 0, stream>>>(agg, bias, nfeat);
}

extern "C" void kernel_launch(void* const* d_in, const int* in_sizes, int n_in,
                              void* d_out, int out_size, void* d_ws, size_t ws_size,
                              hipStream_t stream) {
    const float*     x   = (const float*)d_in[0];
    const long long* ei  = (const long long*)d_in[1];   // int64 [2,E]
    const float*     W1  = (const float*)d_in[2];
    const float*     as1 = (const float*)d_in[3];
    const float*     ad1 = (const float*)d_in[4];
    const float*     b1  = (const float*)d_in[5];
    const float*     W2  = (const float*)d_in[6];
    const float*     as2 = (const float*)d_in[7];
    const float*     ad2 = (const float*)d_in[8];
    const float*     b2  = (const float*)d_in[9];
    const float*     hW1 = (const float*)d_in[10];
    const float*     hb1 = (const float*)d_in[11];
    const float*     hW2 = (const float*)d_in[12];
    const float*     hb2 = (const float*)d_in[13];
    float* out = (float*)d_out;

    const int       N = in_sizes[0] / 64;
    const long long E = (long long)in_sizes[1] / 2;

    // workspace layout
    float*    h_trans = (float*)d_ws;                       // N*64
    float*    agg     = h_trans + (size_t)N * 64;           // N*64
    float*    a_src   = agg + (size_t)N * 64;               // N*4
    float*    a_dst   = a_src + (size_t)N * 4;              // N*4
    unsigned* m_ord   = (unsigned*)(a_dst + (size_t)N * 4); // N*4
    float*    s_sum   = (float*)(m_ord + (size_t)N * 4);    // N*4

    // layer 1: x -> agg
    run_gat_layer(x, W1, as1, ad1, b1, ei, E, N,
                  h_trans, agg, a_src, a_dst, m_ord, s_sum, stream);
    // layer 2: agg -> agg  (GEMM reads agg into h_trans first, then agg is
    // zeroed and re-accumulated — stream ordering makes this safe)
    run_gat_layer(agg, W2, as2, ad2, b2, ei, E, N,
                  h_trans, agg, a_src, a_dst, m_ord, s_sum, stream);
    // MLP head
    mlp_head_k<<<(N + 255) / 256, 256, 0, stream>>>(agg, hW1, hb1, hW2, hb2, out, N);
}
